// GPT_20074677142172
// MI455X (gfx1250) — compile-verified, matrix operand order checked by reading, not run on hardware
//
#include <hip/hip_runtime.h>
#include <hip/hip_bf16.h>
#include <math.h>

// ---------------- model constants ----------------
constexpr int CB   = 2;
constexpr int CS   = 1024;
constexpr int CV   = 8192;
constexpr int CD   = 1024;
constexpr int CH   = 16;
constexpr int CKVH = 4;
constexpr int CHID = 2736;
constexpr int CKV  = 256;         // KVH * HD
constexpr int CM   = CB * CS;     // 2048 tokens
constexpr float CEPS = 1.1920929e-7f;   // float32 eps
constexpr float CSOFTCAP = 30.0f;
constexpr float CLN_ROPE = 9.210340371976184f; // ln(10000)

typedef __attribute__((ext_vector_type(16))) __bf16 v16bf;
typedef __attribute__((ext_vector_type(8)))  __bf16 v8bf;
typedef __attribute__((ext_vector_type(8)))  float  v8f;

__device__ __forceinline__ v8bf pack8(float4 a, float4 b) {
  union { v8bf v; __bf16 e[8]; } u;
  u.e[0] = (__bf16)a.x; u.e[1] = (__bf16)a.y; u.e[2] = (__bf16)a.z; u.e[3] = (__bf16)a.w;
  u.e[4] = (__bf16)b.x; u.e[5] = (__bf16)b.y; u.e[6] = (__bf16)b.z; u.e[7] = (__bf16)b.w;
  return u.v;
}

// =======================================================================
// Generic GEMM: C[M,N] = A[M,K] @ B[N,K]^T  (K-contiguous, f32 in/out)
// bf16 WMMA / f32 accum; 256 thr = 8 waves; tile 128x64; K-step 64;
// double-buffered LDS, one barrier per K-step, loads overlap WMMA.
// =======================================================================
__global__ __launch_bounds__(256) void gemm_xwt_k(const float* __restrict__ A,
                                                  const float* __restrict__ Bw,
                                                  float* __restrict__ C,
                                                  int M, int N, int K) {
  __shared__ alignas(16) __bf16 As[2 * 128 * 64];
  __shared__ alignas(16) __bf16 Bs[2 * 64 * 64];
  const int tid  = threadIdx.x;
  const int wave = tid >> 5;
  const int lane = tid & 31;
  const int rowBase = blockIdx.y * 128;
  const int colBase = blockIdx.x * 64;
  const bool fullTile = (rowBase + 128 <= M) && (colBase + 64 <= N);

  // staging coordinates (K-step = 64)
  const int ra = tid >> 1;            // A row 0..127
  const int ca = (tid & 1) * 32;      // A col 0 or 32
  const int rb = tid >> 2;            // B row 0..63
  const int cb = (tid & 3) * 16;      // B col 0,16,32,48
  const float* aRow = A  + (size_t)(rowBase + ra) * K;
  const float* bRow = Bw + (size_t)(colBase + rb) * K;

  const int r16  = lane & 15;
  const int koff = (lane >> 4) * 8;   // CDNA5 16-bit A/B lane-half K split

  v8f acc[4] = {};
  float4 la[8], lb[4];                // staged global data (next tile)

  auto load_fast = [&](int k0) {
    const float4* a4 = (const float4*)(aRow + k0 + ca);
    const float4* b4 = (const float4*)(bRow + k0 + cb);
    __builtin_prefetch(a4 + 16, 0, 1);   // tile after next
    __builtin_prefetch(b4 + 16, 0, 1);
#pragma unroll
    for (int e = 0; e < 8; ++e) la[e] = a4[e];
#pragma unroll
    for (int e = 0; e < 4; ++e) lb[e] = b4[e];
  };
  auto store_fast = [&](int buf) {
    __bf16* ad = &As[buf * 128 * 64 + ra * 64 + ca];
#pragma unroll
    for (int e = 0; e < 4; ++e) *(v8bf*)(ad + e * 8) = pack8(la[2 * e], la[2 * e + 1]);
    __bf16* bd = &Bs[buf * 64 * 64 + rb * 64 + cb];
#pragma unroll
    for (int e = 0; e < 2; ++e) *(v8bf*)(bd + e * 8) = pack8(lb[2 * e], lb[2 * e + 1]);
  };
  auto stage_slow = [&](int buf, int k0) {
    const int gra = rowBase + ra;
#pragma unroll 4
    for (int e = 0; e < 32; ++e) {
      const int k = k0 + ca + e;
      As[buf * 128 * 64 + ra * 64 + ca + e] =
          (__bf16)((gra < M && k < K) ? aRow[k] : 0.0f);
    }
    const int grb = colBase + rb;
#pragma unroll 4
    for (int e = 0; e < 16; ++e) {
      const int k = k0 + cb + e;
      Bs[buf * 64 * 64 + rb * 64 + cb + e] =
          (__bf16)((grb < N && k < K) ? bRow[k] : 0.0f);
    }
  };
  auto compute = [&](int buf) {
    const __bf16* Ac = &As[buf * 128 * 64];
    const __bf16* Bc = &Bs[buf * 64 * 64];
#pragma unroll
    for (int ks = 0; ks < 2; ++ks) {
      union { v16bf v; v8bf h[2]; } ua;
      ua.h[0] = *(const v8bf*)&Ac[(wave * 16 + r16) * 64 + ks * 32 + koff];
      ua.h[1] = *(const v8bf*)&Ac[(wave * 16 + r16) * 64 + ks * 32 + 16 + koff];
#pragma unroll
      for (int nt = 0; nt < 4; ++nt) {
        union { v16bf v; v8bf h[2]; } ub;
        ub.h[0] = *(const v8bf*)&Bc[(nt * 16 + r16) * 64 + ks * 32 + koff];
        ub.h[1] = *(const v8bf*)&Bc[(nt * 16 + r16) * 64 + ks * 32 + 16 + koff];
        acc[nt] = __builtin_amdgcn_wmma_f32_16x16x32_bf16(
            false, ua.v, false, ub.v, (short)0, acc[nt], false, false);
      }
    }
  };

  const int nIter = (K + 63) / 64;
  // prologue: stage tile 0 into buffer 0
  if (fullTile && 64 <= K) { load_fast(0); store_fast(0); }
  else                     { stage_slow(0, 0); }
  int cur = 0;
  for (int it = 0; it < nIter; ++it) {
    __syncthreads();                       // staged tile `it` visible; buf cur^1 free
    const int nxt = it + 1;
    const bool haveNext = nxt < nIter;
    const bool fastN = haveNext && fullTile && (nxt * 64 + 64 <= K);
    if (fastN) load_fast(nxt * 64);        // issue global loads (wait deferred)
    compute(cur);                          // 8 WMMAs on current LDS buffer
    if (haveNext) {
      if (fastN) store_fast(cur ^ 1);      // s_wait_loadcnt sinks here
      else       stage_slow(cur ^ 1, nxt * 64);
    }
    cur ^= 1;
  }

  // D layout: lane -> n = lane&15 ; vgpr v -> m = v + 8*(lane>=16)
  const int mh = (lane >> 4) * 8;
#pragma unroll
  for (int nt = 0; nt < 4; ++nt) {
    const int n = colBase + nt * 16 + r16;
    if (n >= N) continue;
#pragma unroll
    for (int v = 0; v < 8; ++v) {
      const int m = rowBase + wave * 16 + mh + v;
      if (m < M) C[(size_t)m * N + n] = acc[nt][v];
    }
  }
}

// =======================================================================
// Flash attention per (b,h): Q,O [B,H,S,64]; K,V [B,KVH,S,64]; causal.
// 128 thr = 4 waves, each wave owns 16 query rows; key tiles of 64.
// K row-major in LDS; V transposed ([d][key]) -> contiguous B-fragments.
// =======================================================================
__global__ __launch_bounds__(128) void flash_attn_k(const float* __restrict__ Q,
                                                    const float* __restrict__ Kg,
                                                    const float* __restrict__ Vg,
                                                    float* __restrict__ O) {
  __shared__ alignas(16) __bf16 Ks[64 * 64];
  __shared__ alignas(16) __bf16 Vs[64 * 64];   // transposed: Vs[d*64 + key]
  __shared__ alignas(16) __bf16 Ps[4][16 * 64];

  const int bh  = blockIdx.y;            // b*H + h
  const int h   = bh % CH;
  const int b   = bh / CH;
  const int kvh = h / (CH / CKVH);
  const float* Qp = Q + (size_t)bh * CS * 64;
  float*       Op = O + (size_t)bh * CS * 64;
  const float* Kp = Kg + ((size_t)(b * CKVH + kvh)) * CS * 64;
  const float* Vp = Vg + ((size_t)(b * CKVH + kvh)) * CS * 64;

  const int tid  = threadIdx.x;
  const int wave = tid >> 5;
  const int lane = tid & 31;
  const int r16  = lane & 15;
  const int koff = (lane >> 4) * 8;
  const int mh   = (lane >> 4) * 8;

  // preload Q fragments (K = 64 -> two 32-steps), float4 loads
  v16bf qa[2];
  {
    const int qrow = blockIdx.x * 64 + wave * 16 + r16;
    const float* qsrc = Qp + (size_t)qrow * 64;
#pragma unroll
    for (int ks = 0; ks < 2; ++ks) {
      const float4* lo = (const float4*)(qsrc + ks * 32 + koff);
      const float4* hi = (const float4*)(qsrc + ks * 32 + 16 + koff);
      union { v16bf v; v8bf h2[2]; } u;
      u.h2[0] = pack8(lo[0], lo[1]);
      u.h2[1] = pack8(hi[0], hi[1]);
      qa[ks] = u.v;
    }
  }

  float m_i[8], l_i[8];
  v8f acc[4] = {};
#pragma unroll
  for (int v = 0; v < 8; ++v) { m_i[v] = -1e30f; l_i[v] = 0.0f; }
  const float scale = 0.125f; // 1/sqrt(64)

  const int ntiles = blockIdx.x + 1;   // causal: key tile <= query tile
  for (int kt = 0; kt < ntiles; ++kt) {
    { // stage K (row-major) and V (transposed), f32 -> bf16, float4 loads
      const int r  = tid >> 1;          // key row 0..63
      const int c0 = (tid & 1) * 32;    // 32 d-columns per thread
      const float4* k4 = (const float4*)(Kp + (size_t)(kt * 64 + r) * 64 + c0);
      const float4* v4 = (const float4*)(Vp + (size_t)(kt * 64 + r) * 64 + c0);
#pragma unroll
      for (int q8 = 0; q8 < 4; ++q8) {
        float4 a = k4[2 * q8], bq = k4[2 * q8 + 1];
        *(v8bf*)&Ks[r * 64 + c0 + q8 * 8] = pack8(a, bq);
      }
#pragma unroll
      for (int q8 = 0; q8 < 4; ++q8) {
        float4 a = v4[2 * q8], bq = v4[2 * q8 + 1];
        const int d0 = c0 + q8 * 8;
        Vs[(d0 + 0) * 64 + r] = (__bf16)a.x;
        Vs[(d0 + 1) * 64 + r] = (__bf16)a.y;
        Vs[(d0 + 2) * 64 + r] = (__bf16)a.z;
        Vs[(d0 + 3) * 64 + r] = (__bf16)a.w;
        Vs[(d0 + 4) * 64 + r] = (__bf16)bq.x;
        Vs[(d0 + 5) * 64 + r] = (__bf16)bq.y;
        Vs[(d0 + 6) * 64 + r] = (__bf16)bq.z;
        Vs[(d0 + 7) * 64 + r] = (__bf16)bq.w;
      }
    }
    __syncthreads();

    // scores S = Q K^T (4 key sub-tiles of 16)
    v8f sc[4];
#pragma unroll
    for (int nt = 0; nt < 4; ++nt) {
      v8f z = {};
#pragma unroll
      for (int ks = 0; ks < 2; ++ks) {
        union { v16bf v; v8bf h2[2]; } u;
        u.h2[0] = *(const v8bf*)&Ks[(nt * 16 + r16) * 64 + ks * 32 + koff];
        u.h2[1] = *(const v8bf*)&Ks[(nt * 16 + r16) * 64 + ks * 32 + 16 + koff];
        z = __builtin_amdgcn_wmma_f32_16x16x32_bf16(
            false, qa[ks], false, u.v, (short)0, z, false, false);
      }
      sc[nt] = z;
    }

    // causal mask + per-row max
    float rowmax[8];
#pragma unroll
    for (int v = 0; v < 8; ++v) rowmax[v] = -1e30f;
#pragma unroll
    for (int nt = 0; nt < 4; ++nt) {
      const int key = kt * 64 + nt * 16 + r16;
#pragma unroll
      for (int v = 0; v < 8; ++v) {
        const int qr = blockIdx.x * 64 + wave * 16 + mh + v;
        float s = sc[nt][v] * scale;
        if (key > qr) s = -1e30f;
        sc[nt][v] = s;
        rowmax[v] = fmaxf(rowmax[v], s);
      }
    }
#pragma unroll
    for (int v = 0; v < 8; ++v)
      for (int m = 1; m < 16; m <<= 1)
        rowmax[v] = fmaxf(rowmax[v], __shfl_xor(rowmax[v], m, 32));

    // online softmax update
    float rescale[8], rowsum[8];
#pragma unroll
    for (int v = 0; v < 8; ++v) {
      const float nm = fmaxf(m_i[v], rowmax[v]);
      rescale[v] = __expf(m_i[v] - nm);
      m_i[v] = nm;
      rowsum[v] = 0.0f;
    }
#pragma unroll
    for (int nt = 0; nt < 4; ++nt)
#pragma unroll
      for (int v = 0; v < 8; ++v) {
        const float p = __expf(sc[nt][v] - m_i[v]);
        sc[nt][v] = p;
        rowsum[v] += p;
      }
#pragma unroll
    for (int v = 0; v < 8; ++v) {
      for (int m = 1; m < 16; m <<= 1) rowsum[v] += __shfl_xor(rowsum[v], m, 32);
      l_i[v] = l_i[v] * rescale[v] + rowsum[v];
    }
#pragma unroll
    for (int nt = 0; nt < 4; ++nt)
#pragma unroll
      for (int v = 0; v < 8; ++v) acc[nt][v] *= rescale[v];

    // P (D-layout) -> LDS -> A-layout fragments
#pragma unroll
    for (int nt = 0; nt < 4; ++nt)
#pragma unroll
      for (int v = 0; v < 8; ++v)
        Ps[wave][(mh + v) * 64 + nt * 16 + r16] = (__bf16)sc[nt][v];

    // O += P @ V  (V transposed in LDS -> contiguous B-fragment loads)
#pragma unroll
    for (int ks = 0; ks < 2; ++ks) {
      union { v16bf v; v8bf h2[2]; } pu;
      pu.h2[0] = *(const v8bf*)&Ps[wave][r16 * 64 + ks * 32 + koff];
      pu.h2[1] = *(const v8bf*)&Ps[wave][r16 * 64 + ks * 32 + 16 + koff];
#pragma unroll
      for (int dt = 0; dt < 4; ++dt) {
        const int col = dt * 16 + r16;     // d column
        union { v16bf v; v8bf h2[2]; } vu;
        vu.h2[0] = *(const v8bf*)&Vs[col * 64 + ks * 32 + koff];
        vu.h2[1] = *(const v8bf*)&Vs[col * 64 + ks * 32 + 16 + koff];
        acc[dt] = __builtin_amdgcn_wmma_f32_16x16x32_bf16(
            false, pu.v, false, vu.v, (short)0, acc[dt], false, false);
      }
    }
    __syncthreads();
  }

  // normalize + store
#pragma unroll
  for (int dt = 0; dt < 4; ++dt)
#pragma unroll
    for (int v = 0; v < 8; ++v) {
      const int m = blockIdx.x * 64 + wave * 16 + mh + v;
      Op[(size_t)m * 64 + dt * 16 + r16] = acc[dt][v] / l_i[v];
    }
}

// =======================================================================
// Elementwise / normalization helpers (float4 where profitable)
// =======================================================================
__global__ __launch_bounds__(256) void rmsnorm_k(const float* __restrict__ x,
                                                 const float* __restrict__ w,
                                                 float* __restrict__ y, int C) {
  const int row = blockIdx.x;
  const float* xr = x + (size_t)row * C;
  float* yr = y + (size_t)row * C;
  __shared__ float red[8];
  float ss = 0.0f;
  for (int c = threadIdx.x; c < C; c += 256) { const float v = xr[c]; ss += v * v; }
  for (int m = 16; m; m >>= 1) ss += __shfl_xor(ss, m, 32);
  if ((threadIdx.x & 31) == 0) red[threadIdx.x >> 5] = ss;
  __syncthreads();
  float tot = 0.0f;
#pragma unroll
  for (int i = 0; i < 8; ++i) tot += red[i];
  const float rs = rsqrtf(tot / (float)C + CEPS);
  for (int c = threadIdx.x; c < C; c += 256)
    yr[c] = xr[c] * rs * (w ? w[c] : 1.0f);
}

__global__ __launch_bounds__(256) void embed_gather_k(const int* __restrict__ idx,
                                                      const float* __restrict__ emb,
                                                      float* __restrict__ x, size_t total4) {
  const size_t i4 = (size_t)blockIdx.x * 256 + threadIdx.x;
  if (i4 >= total4) return;
  const int d4 = (int)(i4 % (CD / 4));
  const size_t t = i4 / (CD / 4);
  ((float4*)x)[i4] = ((const float4*)(emb + (size_t)idx[t] * CD))[d4];
}

__global__ __launch_bounds__(256) void resid_mix_k(float* __restrict__ x,
                                                   const float* __restrict__ x0,
                                                   const float* __restrict__ m0,
                                                   const float* __restrict__ m1,
                                                   size_t total4) {
  const size_t i4 = (size_t)blockIdx.x * 256 + threadIdx.x;
  if (i4 >= total4) return;
  const int d4 = (int)(i4 % (CD / 4));
  float4 a = ((float4*)x)[i4], b = ((const float4*)x0)[i4];
  float4 w0 = ((const float4*)m0)[d4], w1 = ((const float4*)m1)[d4];
  a.x = w0.x * a.x + w1.x * b.x;  a.y = w0.y * a.y + w1.y * b.y;
  a.z = w0.z * a.z + w1.z * b.z;  a.w = w0.w * a.w + w1.w * b.w;
  ((float4*)x)[i4] = a;
}

__global__ __launch_bounds__(256) void axpy_scale_k(float* __restrict__ x,
                                                    const float* __restrict__ y,
                                                    const float* __restrict__ s,
                                                    size_t total4) {
  const size_t i4 = (size_t)blockIdx.x * 256 + threadIdx.x;
  if (i4 >= total4) return;
  const int d4 = (int)(i4 % (CD / 4));
  float4 a = ((float4*)x)[i4], b = ((const float4*)y)[i4];
  float4 w = ((const float4*)s)[d4];
  a.x += w.x * b.x; a.y += w.y * b.y; a.z += w.z * b.z; a.w += w.w * b.w;
  ((float4*)x)[i4] = a;
}

__global__ __launch_bounds__(256) void skip_add_k(float* __restrict__ x,
                                                  const float* __restrict__ sk,
                                                  const float* __restrict__ sw,
                                                  const float* __restrict__ sg,
                                                  size_t total4) {
  const size_t i4 = (size_t)blockIdx.x * 256 + threadIdx.x;
  if (i4 >= total4) return;
  const int d4 = (int)(i4 % (CD / 4));
  float4 a = ((float4*)x)[i4], b = ((const float4*)sk)[i4];
  float4 w = ((const float4*)sw)[d4], g = ((const float4*)sg)[d4];
  a.x += w.x * (1.0f / (1.0f + expf(-g.x))) * b.x;
  a.y += w.y * (1.0f / (1.0f + expf(-g.y))) * b.y;
  a.z += w.z * (1.0f / (1.0f + expf(-g.z))) * b.z;
  a.w += w.w * (1.0f / (1.0f + expf(-g.w))) * b.w;
  ((float4*)x)[i4] = a;
}

__global__ __launch_bounds__(256) void silu_mul_k(float* __restrict__ h1,
                                                  const float* __restrict__ h2,
                                                  size_t total4) {
  const size_t i4 = (size_t)blockIdx.x * 256 + threadIdx.x;
  if (i4 >= total4) return;
  float4 a = ((float4*)h1)[i4], g = ((const float4*)h2)[i4];
  a.x *= g.x / (1.0f + expf(-g.x));
  a.y *= g.y / (1.0f + expf(-g.y));
  a.z *= g.z / (1.0f + expf(-g.z));
  a.w *= g.w / (1.0f + expf(-g.w));
  ((float4*)h1)[i4] = a;
}

__global__ __launch_bounds__(256) void dwconv_k(const float* __restrict__ x,
                                                const float* __restrict__ w,
                                                float* __restrict__ y,
                                                int C, size_t total4) {
  const size_t i4 = (size_t)blockIdx.x * 256 + threadIdx.x;
  if (i4 >= total4) return;
  const int C4 = C / 4;
  const int c4 = (int)(i4 % C4);
  const int s  = (int)((i4 / C4) % CS);
  float4 w0 = ((const float4*)w)[c4];
  float4 w1 = ((const float4*)(w + C))[c4];
  float4 w2 = ((const float4*)(w + 2 * C))[c4];
  float4 x2 = ((const float4*)x)[i4];
  float4 r;
  r.x = w2.x * x2.x; r.y = w2.y * x2.y; r.z = w2.z * x2.z; r.w = w2.w * x2.w;
  if (s >= 1) {
    float4 x1 = ((const float4*)x)[i4 - C4];
    r.x += w1.x * x1.x; r.y += w1.y * x1.y; r.z += w1.z * x1.z; r.w += w1.w * x1.w;
  }
  if (s >= 2) {
    float4 x0 = ((const float4*)x)[i4 - 2 * (size_t)C4];
    r.x += w0.x * x0.x; r.y += w0.y * x0.y; r.z += w0.z * x0.z; r.w += w0.w * x0.w;
  }
  ((float4*)y)[i4] = r;
}

// rmsnorm(no weight) + RoPE + transpose [b,s,h,64] -> [b,h,s,64]; wave per row
__global__ __launch_bounds__(256) void rope_norm_tr_k(const float* __restrict__ in,
                                                      float* __restrict__ out,
                                                      int nh, int nrows) {
  const int row = blockIdx.x * 8 + (threadIdx.x >> 5);
  if (row >= nrows) return;
  const int lane = threadIdx.x & 31;
  const int h  = row % nh;
  const int bs = row / nh;
  const int s  = bs % CS;
  const int b  = bs / CS;
  const float* src = in + (size_t)row * 64;
  float x1 = src[lane], x2 = src[lane + 32];
  float ss = x1 * x1 + x2 * x2;
  for (int m = 1; m < 32; m <<= 1) ss += __shfl_xor(ss, m, 32);
  const float rs = rsqrtf(ss / 64.0f + CEPS);
  const float y1 = x1 * rs, y2 = x2 * rs;
  const float theta = (float)s * __expf(-(float)lane * (1.0f / 32.0f) * CLN_ROPE);
  const float c = __cosf(theta), sn = __sinf(theta);
  float* dst = out + ((size_t)(b * nh + h) * CS + s) * 64;
  dst[lane]      =  y1 * c + y2 * sn;
  dst[lane + 32] = -y1 * sn + y2 * c;
}

// [b,s,h,64] -> [b,h,s,64]  (float4 along hd)
__global__ __launch_bounds__(256) void tr_heads_k(const float* __restrict__ in,
                                                  float* __restrict__ out,
                                                  int nh, size_t total4) {
  const size_t i4 = (size_t)blockIdx.x * 256 + threadIdx.x;
  if (i4 >= total4) return;
  const int hd4 = (int)(i4 & 15);
  const size_t r = i4 >> 4;
  const int s = (int)(r % CS);
  const size_t r2 = r / CS;
  const int h = (int)(r2 % nh);
  const int b = (int)(r2 / nh);
  ((float4*)out)[i4] =
      ((const float4*)in)[(((size_t)(b * CS + s) * nh + h) << 4) + hd4];
}

// [b,h,s,64] -> [b,s,h,64]  (float4 along hd)
__global__ __launch_bounds__(256) void tr_back_k(const float* __restrict__ in,
                                                 float* __restrict__ out,
                                                 int nh, size_t total4) {
  const size_t i4 = (size_t)blockIdx.x * 256 + threadIdx.x;
  if (i4 >= total4) return;
  const int hd4 = (int)(i4 & 15);
  const size_t r = i4 >> 4;
  const int h = (int)(r % nh);
  const size_t r2 = r / nh;
  const int s = (int)(r2 % CS);
  const int b = (int)(r2 / CS);
  ((float4*)out)[i4] =
      ((const float4*)in)[(((size_t)(b * nh + h) * CS + s) << 4) + hd4];
}

// q' = einsum('hg,bgsd->bhsd')  (float4 along hd)
__global__ __launch_bounds__(256) void qgain_k(const float* __restrict__ in,
                                               const float* __restrict__ gain,
                                               float* __restrict__ out,
                                               size_t total4) {
  const size_t i4 = (size_t)blockIdx.x * 256 + threadIdx.x;
  if (i4 >= total4) return;
  const int hd4 = (int)(i4 & 15);
  const size_t r = i4 >> 4;
  const int s = (int)(r % CS);
  const size_t r2 = r / CS;
  const int h = (int)(r2 % CH);
  const int b = (int)(r2 / CH);
  float4 a = {0.f, 0.f, 0.f, 0.f};
#pragma unroll
  for (int g = 0; g < CH; ++g) {
    const float w = gain[h * CH + g];
    float4 q = ((const float4*)in)[(((size_t)(b * CH + g) * CS + s) << 4) + hd4];
    a.x += w * q.x; a.y += w * q.y; a.z += w * q.z; a.w += w * q.w;
  }
  ((float4*)out)[i4] = a;
}

__global__ __launch_bounds__(256) void gate_k(const float* __restrict__ xn,
                                              const float* __restrict__ bg,
                                              float* __restrict__ gate) {
  const int t = blockIdx.x;
  __shared__ float red[8];
  float s = 0.0f;
  for (int c = threadIdx.x; c < CD; c += 256) s += xn[(size_t)t * CD + c] * bg[c];
  for (int m = 16; m; m >>= 1) s += __shfl_xor(s, m, 32);
  if ((threadIdx.x & 31) == 0) red[threadIdx.x >> 5] = s;
  __syncthreads();
  if (threadIdx.x == 0) {
    float tot = 0.0f;
#pragma unroll
    for (int i = 0; i < 8; ++i) tot += red[i];
    gate[t] = 1.0f / (1.0f + expf(-tot));
  }
}

__global__ __launch_bounds__(256) void logits_epilogue_k(float* __restrict__ logits,
                                                         const float* __restrict__ ub,
                                                         const float* __restrict__ gate,
                                                         const int* __restrict__ idx,
                                                         const float* __restrict__ bh,
                                                         const float* __restrict__ bs,
                                                         size_t total4) {
  const size_t i4 = (size_t)blockIdx.x * 256 + threadIdx.x;
  if (i4 >= total4) return;
  const int v4 = (int)(i4 % (CV / 4));
  const size_t t = i4 / (CV / 4);
  const float g = gate[t];
  float4 lg = ((float4*)logits)[i4];
  float4 u  = ((const float4*)ub)[v4];
  float4 sc = ((const float4*)bs)[v4];
  float4 bb = ((const float4*)(bh + (size_t)idx[t] * CV))[v4];
  lg.x = CSOFTCAP * tanhf((lg.x + u.x + g * bb.x * sc.x) / CSOFTCAP);
  lg.y = CSOFTCAP * tanhf((lg.y + u.y + g * bb.y * sc.y) / CSOFTCAP);
  lg.z = CSOFTCAP * tanhf((lg.z + u.z + g * bb.z * sc.z) / CSOFTCAP);
  lg.w = CSOFTCAP * tanhf((lg.w + u.w + g * bb.w * sc.w) / CSOFTCAP);
  ((float4*)logits)[i4] = lg;
}

// =======================================================================
// Orchestration
// =======================================================================
extern "C" void kernel_launch(void* const* d_in, const int* in_sizes, int n_in,
                              void* d_out, int out_size, void* d_ws, size_t ws_size,
                              hipStream_t stream) {
  const int*   idx              = (const int*)  d_in[0];
  const float* tok_emb          = (const float*)d_in[1];
  const float* embed_norm_w     = (const float*)d_in[2];
  const float* resid_mix        = (const float*)d_in[3];
  const float* attn_norm_w      = (const float*)d_in[4];
  const float* mlp_norm_w       = (const float*)d_in[5];
  const float* attn_scale       = (const float*)d_in[6];
  const float* mlp_scale        = (const float*)d_in[7];
  const float* Wq               = (const float*)d_in[8];
  const float* Wk               = (const float*)d_in[9];
  const float* Wv               = (const float*)d_in[10];
  const float* q_conv_w         = (const float*)d_in[11];
  const float* k_conv_w         = (const float*)d_in[12];
  const float* v_conv_w         = (const float*)d_in[13];
  const float* q_gain           = (const float*)d_in[14];
  const float* post_attn_norm_w = (const float*)d_in[15];
  const float* Wo               = (const float*)d_in[16];
  const float* Wfc              = (const float*)d_in[17];
  const float* Wgate            = (const float*)d_in[18];
  const float* h_conv_w         = (const float*)d_in[19];
  const float* h_norm_w         = (const float*)d_in[20];
  const float* Wmproj           = (const float*)d_in[21];
  const float* skip_weights     = (const float*)d_in[22];
  const float* skip_gates       = (const float*)d_in[23];
  const float* final_norm_w     = (const float*)d_in[24];
  const float* lm_head_w        = (const float*)d_in[25];
  const float* bigram_head      = (const float*)d_in[26];
  const float* bigram_scale     = (const float*)d_in[27];
  const float* bigram_gate      = (const float*)d_in[28];
  const float* unigram_bias     = (const float*)d_in[29];
  float* out = (float*)d_out;

  // ---- workspace carve ----
  char* wsb = (char*)d_ws;
  size_t off = 0;
  auto alloc = [&](size_t nf) -> float* {
    float* p = (float*)(wsb + off);
    off += nf * sizeof(float);
    off = (off + 255) & ~(size_t)255;
    return p;
  };
  const size_t totXD = (size_t)CM * CD;
  const size_t totKV = (size_t)CM * CKV;
  const size_t totH  = (size_t)CM * CHID;

  float* X  = alloc(totXD);
  float* X0 = alloc(totXD);
  float* XN = alloc(totXD);
  float* QL = alloc(totXD);
  float* QC = alloc(totXD);
  float* KL = alloc(totKV);
  float* KC = alloc(totKV);
  float* VL = alloc(totKV);
  float* VC = alloc(totKV);
  float* QT = alloc(totXD);
  float* QG = alloc(totXD);
  float* KT = alloc(totKV);
  float* VT = alloc(totKV);
  float* YT = alloc(totXD);
  float* Y  = alloc(totXD);
  float* YN = alloc(totXD);
  float* T1 = alloc(totXD);
  float* H1 = alloc(totH);
  float* H2 = alloc(totH);
  float* SK[4];
  for (int i = 0; i < 4; ++i) SK[i] = alloc(totXD);
  float* GATE = alloc(CM);

  const dim3 b256(256);
  auto ew = [](size_t total) { return dim3((unsigned)((total + 255) / 256)); };
  auto gemm = [&](const float* A, const float* Bw, float* C, int M_, int N_, int K_) {
    dim3 g((N_ + 63) / 64, (M_ + 127) / 128);
    gemm_xwt_k<<<g, 256, 0, stream>>>(A, Bw, C, M_, N_, K_);
  };

  // ---- embedding ----
  embed_gather_k<<<ew(totXD / 4), b256, 0, stream>>>(idx, tok_emb, QL, totXD / 4);
  rmsnorm_k<<<CM, 256, 0, stream>>>(QL, embed_norm_w, X, CD);
  hipMemcpyAsync(X0, X, totXD * sizeof(float), hipMemcpyDeviceToDevice, stream);

  auto block = [&](int i) {
    // residual mix
    resid_mix_k<<<ew(totXD / 4), b256, 0, stream>>>(
        X, X0, resid_mix + (size_t)(i * 2) * CD, resid_mix + (size_t)(i * 2 + 1) * CD,
        totXD / 4);

    // ---- attention ----
    rmsnorm_k<<<CM, 256, 0, stream>>>(X, attn_norm_w + (size_t)i * CD, XN, CD);
    gemm(XN, Wq + (size_t)i * CD * CD, QL, CM, CD, CD);
    gemm(XN, Wk + (size_t)i * CKV * CD, KL, CM, CKV, CD);
    gemm(XN, Wv + (size_t)i * CKV * CD, VL, CM, CKV, CD);
    dwconv_k<<<ew(totXD / 4), b256, 0, stream>>>(QL, q_conv_w + (size_t)i * 3 * CD, QC, CD, totXD / 4);
    dwconv_k<<<ew(totKV / 4), b256, 0, stream>>>(KL, k_conv_w + (size_t)i * 3 * CKV, KC, CKV, totKV / 4);
    dwconv_k<<<ew(totKV / 4), b256, 0, stream>>>(VL, v_conv_w + (size_t)i * 3 * CKV, VC, CKV, totKV / 4);
    rope_norm_tr_k<<<(CB * CS * CH) / 8, 256, 0, stream>>>(QC, QT, CH, CB * CS * CH);
    rope_norm_tr_k<<<(CB * CS * CKVH) / 8, 256, 0, stream>>>(KC, KT, CKVH, CB * CS * CKVH);
    tr_heads_k<<<ew(totKV / 4), b256, 0, stream>>>(VC, VT, CKVH, totKV / 4);
    qgain_k<<<ew(totXD / 4), b256, 0, stream>>>(QT, q_gain + (size_t)i * CH * CH, QG, totXD / 4);
    dim3 fg(CS / 64, CB * CH);
    flash_attn_k<<<fg, 128, 0, stream>>>(QG, KT, VT, YT);
    tr_back_k<<<ew(totXD / 4), b256, 0, stream>>>(YT, Y, CH, totXD / 4);
    rmsnorm_k<<<CM, 256, 0, stream>>>(Y, post_attn_norm_w + (size_t)i * CD, YN, CD);
    gemm(YN, Wo + (size_t)i * CD * CD, T1, CM, CD, CD);
    axpy_scale_k<<<ew(totXD / 4), b256, 0, stream>>>(X, T1, attn_scale + (size_t)i * CD, totXD / 4);

    // ---- MLP ----
    rmsnorm_k<<<CM, 256, 0, stream>>>(X, mlp_norm_w + (size_t)i * CD, XN, CD);
    gemm(XN, Wfc + (size_t)i * CHID * CD, H1, CM, CHID, CD);
    gemm(XN, Wgate + (size_t)i * CHID * CD, H2, CM, CHID, CD);
    silu_mul_k<<<ew(totH / 4), b256, 0, stream>>>(H1, H2, totH / 4);
    dwconv_k<<<ew(totH / 4), b256, 0, stream>>>(H1, h_conv_w + (size_t)i * 3 * CHID, H2, CHID, totH / 4);
    rmsnorm_k<<<CM, 256, 0, stream>>>(H2, h_norm_w + (size_t)i * CHID, H2, CHID);
    gemm(H2, Wmproj + (size_t)i * CD * CHID, T1, CM, CD, CHID);
    axpy_scale_k<<<ew(totXD / 4), b256, 0, stream>>>(X, T1, mlp_scale + (size_t)i * CD, totXD / 4);
  };

  // encoder half: run block, save skip
  for (int i = 0; i < 4; ++i) {
    block(i);
    hipMemcpyAsync(SK[i], X, totXD * sizeof(float), hipMemcpyDeviceToDevice, stream);
  }
  // decoder half: consume skip (LIFO), run block
  for (int j = 0; j < 4; ++j) {
    skip_add_k<<<ew(totXD / 4), b256, 0, stream>>>(
        X, SK[3 - j], skip_weights + (size_t)j * CD, skip_gates + (size_t)j * CD, totXD / 4);
    block(4 + j);
  }

  // ---- head ----
  rmsnorm_k<<<CM, 256, 0, stream>>>(X, final_norm_w, XN, CD);
  gate_k<<<CM, 256, 0, stream>>>(XN, bigram_gate, GATE);
  gemm(XN, lm_head_w, out, CM, CV, CD);
  const size_t totL4 = (size_t)CM * CV / 4;
  logits_epilogue_k<<<ew(totL4), b256, 0, stream>>>(
      out, unigram_bias, GATE, idx, bigram_head, bigram_scale, totL4);
}
